// SelfAttention_67327907332663
// MI455X (gfx1250) — matrix operand order repaired and checked by reading, not compiled, not run on hardware
//
#include <hip/hip_runtime.h>
#include <hip/hip_bf16.h>

typedef __attribute__((ext_vector_type(16))) _Float16 v16h;
typedef __attribute__((ext_vector_type(8)))  _Float16 v8h;
typedef __attribute__((ext_vector_type(8)))  float    v8f;

#define C_DIM 256
#define NTOK  32768      // B*H*W = 8*64*64
#define SEQ   4096       // H*W tokens per batch
#define NBATCH 8
#define ATT_SCALE 0.0625f  // 1/sqrt(256)

// ---------------------------------------------------------------- WMMA helpers

__device__ __forceinline__ v8f wmma16(v16h a, v16h b, v8f c) {
  return __builtin_amdgcn_wmma_f32_16x16x32_f16(false, a, false, b, (short)0, c,
                                                false, false);
}

// A fragment (16x32 f16, MxK): lane row = lane&15, g = lane>>4.
// elements 0..7  -> K = kBase + 8g + j
// elements 8..15 -> K = kBase + 16 + 8g + (j-8)
__device__ __forceinline__ v16h load_a16(const _Float16* base, int ld, int row,
                                         int kBase, int lane) {
  const int g = lane >> 4;
  const _Float16* p = base + (size_t)row * ld + kBase + 8 * g;
  union { v16h v; v8h h[2]; } u;
  u.h[0] = *(const v8h*)(p);
  u.h[1] = *(const v8h*)(p + 16);
  return u.v;
}

// B fragment (32x16 f16, KxN): lane col passed in; elements 0..7 -> K = kBase + 8g + j,
// elements 8..15 -> K = kBase + 16 + 8g + (j-8)
__device__ __forceinline__ v16h load_b16(const _Float16* base, int ld, int col,
                                         int kBase, int lane) {
  const int g = lane >> 4;
  const _Float16* p = base + (size_t)col * ld + kBase + 8 * g;
  union { v16h v; v8h h[2]; } u;
  u.h[0] = *(const v8h*)(p);
  u.h[1] = *(const v8h*)(p + 16);
  return u.v;
}

// Flat pointers to __shared__ lower as (LDS aperture | byte offset); the low
// 32 bits are the workgroup-relative LDS address the async-DMA VGPR expects.
__device__ __forceinline__ unsigned lds_off(const void* p) {
  return (unsigned)(unsigned long long)p;
}

// gfx1250 async global->LDS copy, 16 bytes per lane (tracked by ASYNCcnt).
__device__ __forceinline__ void async_to_lds_b128(unsigned lds_byte_off,
                                                  unsigned long long gaddr) {
  asm volatile("global_load_async_to_lds_b128 %0, %1, off"
               :: "v"(lds_byte_off), "v"(gaddr) : "memory");
}

__device__ __forceinline__ void wait_async0() {
  asm volatile("s_wait_asynccnt 0" ::: "memory");
}

// ---------------------------------------------------------------- LayerNorm

__global__ void __launch_bounds__(256) ln_kernel(const float* __restrict__ x,
                                                 const float* __restrict__ gamma,
                                                 const float* __restrict__ beta,
                                                 float* __restrict__ xn,
                                                 _Float16* __restrict__ x16) {
  const int lane  = threadIdx.x & 31;
  const int token = (blockIdx.x * blockDim.x + threadIdx.x) >> 5;
  const float* row = x + (size_t)token * C_DIM;

  const float4* rp = (const float4*)(row + lane * 8);
  float4 a0 = rp[0], a1 = rp[1];
  float v[8] = {a0.x, a0.y, a0.z, a0.w, a1.x, a1.y, a1.z, a1.w};

  float s = 0.f;
#pragma unroll
  for (int j = 0; j < 8; ++j) s += v[j];
#pragma unroll
  for (int m = 1; m < 32; m <<= 1) s += __shfl_xor(s, m, 32);
  const float mean = s * (1.0f / 256.0f);

  float q = 0.f;
#pragma unroll
  for (int j = 0; j < 8; ++j) { float d = v[j] - mean; q += d * d; }
#pragma unroll
  for (int m = 1; m < 32; m <<= 1) q += __shfl_xor(q, m, 32);
  const float rs = rsqrtf(q * (1.0f / 256.0f) + 1e-3f);

  float o[8];
  v8h h;
#pragma unroll
  for (int j = 0; j < 8; ++j) {
    const int c = lane * 8 + j;
    o[j] = (v[j] - mean) * rs * gamma[c] + beta[c];
    h[j] = (_Float16)o[j];
  }
  float4* wp = (float4*)(xn + (size_t)token * C_DIM + lane * 8);
  wp[0] = make_float4(o[0], o[1], o[2], o[3]);
  wp[1] = make_float4(o[4], o[5], o[6], o[7]);
  *(v8h*)(x16 + (size_t)token * C_DIM + lane * 8) = h;
}

// ------------------------------------------------- weight transpose f32 -> f16

__global__ void wt_kernel(const float* __restrict__ w, _Float16* __restrict__ wT) {
  const int idx = blockIdx.x * blockDim.x + threadIdx.x;   // 65536 elements
  const int i = idx >> 8;    // input dim
  const int o = idx & 255;   // output dim
  wT[(size_t)o * C_DIM + i] = (_Float16)w[(size_t)i * C_DIM + o];
}

// ---------------------------------------------------------------- QKV GEMM

__global__ void __launch_bounds__(256) qkv_kernel(
    const _Float16* __restrict__ x16,
    const _Float16* __restrict__ wTq, const _Float16* __restrict__ wTk,
    const _Float16* __restrict__ wTv,
    const float* __restrict__ bq, const float* __restrict__ bk,
    const float* __restrict__ bv,
    _Float16* __restrict__ q16, _Float16* __restrict__ k16,
    _Float16* __restrict__ vT16) {
  const int lane = threadIdx.x & 31;
  const int l15 = lane & 15, g = lane >> 4;
  const int wid = (blockIdx.x * blockDim.x + threadIdx.x) >> 5;
  const int mTile = wid / 12;
  const int rem = wid % 12;
  const int mat = rem >> 2;            // 0=Q 1=K 2=V
  const int nBase = (rem & 3) * 64;
  const int rowBase = mTile * 16;

  const _Float16* wT  = (mat == 0) ? wTq : (mat == 1) ? wTk : wTv;
  const float*    bias = (mat == 0) ? bq : (mat == 1) ? bk : bv;

  v8f acc[4] = {};
  for (int k8 = 0; k8 < 8; ++k8) {
    const int kB = k8 * 32;
    v16h a = load_a16(x16, C_DIM, rowBase + l15, kB, lane);
#pragma unroll
    for (int t = 0; t < 4; ++t) {
      v16h b = load_b16(wT, C_DIM, nBase + 16 * t + l15, kB, lane);
      acc[t] = wmma16(a, b, acc[t]);
    }
  }

#pragma unroll
  for (int t = 0; t < 4; ++t) {
    const int col = nBase + 16 * t + l15;
    const float bcol = bias[col];
#pragma unroll
    for (int r = 0; r < 8; ++r) {
      const int token = rowBase + 8 * g + r;
      const float val = acc[t][r] + bcol;
      if (mat == 0) {
        q16[(size_t)token * C_DIM + col] = (_Float16)(val * ATT_SCALE);
      } else if (mat == 1) {
        k16[(size_t)token * C_DIM + col] = (_Float16)val;
      } else {
        const int b  = token >> 12;      // batch
        const int tl = token & 4095;     // token within batch
        vT16[(size_t)b * C_DIM * SEQ + (size_t)col * SEQ + tl] = (_Float16)val;
      }
    }
  }
}

// --------------------------------------------------- fused flash attention
//
// One wave owns a 16-query tile (Q fragments in registers); the 4 waves of a
// block stream the SAME K/V tiles of one batch. Each 32-key step's K tile
// (32x256 f16) and V tile (256x32 f16, channel-major) are staged once per
// block via gfx1250 async global->LDS DMA, DOUBLE-BUFFERED so the DMA for
// step s+1 overlaps the WMMA/softmax compute of step s.

__global__ void __launch_bounds__(128, 1)
attn_kernel(const _Float16* __restrict__ q16,
            const _Float16* __restrict__ k16,
            const _Float16* __restrict__ vT16,
            _Float16* __restrict__ o16) {
  __shared__ __align__(32) _Float16 ldsK[2][32 * C_DIM];  // 2 x 16 KB
  __shared__ __align__(32) _Float16 ldsV[2][C_DIM * 32];  // 2 x 16 KB

  const int tid = threadIdx.x;
  const int lane = tid & 31;
  const int l15 = lane & 15, g = lane >> 4;
  const int wIB = tid >> 5;
  const int wid = blockIdx.x * 4 + wIB;
  const int batch = wid >> 8;            // 256 query tiles per batch
  const int qTile = wid & 255;
  const int qRow0 = batch * SEQ + qTile * 16;   // global token of first query row

  // Preload this wave's Q tile as 8 A-fragments (64 VGPRs), scale pre-folded.
  v16h qf[8];
#pragma unroll
  for (int k8 = 0; k8 < 8; ++k8)
    qf[k8] = load_a16(q16, C_DIM, qRow0 + l15, k8 * 32, lane);

  v8f o[16] = {};
  float mrow[8], lrow[8];
#pragma unroll
  for (int r = 0; r < 8; ++r) { mrow[r] = -3.0e38f; lrow[r] = 0.0f; }

  const _Float16* kbase = k16 + (size_t)batch * SEQ * C_DIM;
  const _Float16* vbase = vT16 + (size_t)batch * (size_t)C_DIM * SEQ;

  // Cooperative async stage of one K tile (contiguous 16 KB) and one V tile
  // (256 channel rows x 64 B) into LDS buffer `buf`.
  auto issue = [&](int buf, int kv0) {
    const unsigned kOff = lds_off(ldsK[buf]);
    const unsigned vOff = lds_off(ldsV[buf]);
    const unsigned long long kg =
        (unsigned long long)(kbase + (size_t)kv0 * C_DIM);
#pragma unroll
    for (int i = 0; i < 8; ++i) {
      const int c = tid + 128 * i;                       // 16-byte chunk id
      async_to_lds_b128(kOff + c * 16, kg + (unsigned long long)c * 16);
    }
#pragma unroll
    for (int i = 0; i < 8; ++i) {
      const int c = tid + 128 * i;
      const int ch = c >> 2, j = c & 3;
      async_to_lds_b128(vOff + c * 16,
          (unsigned long long)(vbase + (size_t)ch * SEQ + kv0 + j * 8));
    }
  };

  issue(0, 0);  // prologue

  for (int step = 0; step < SEQ / 32; ++step) {
    const int cur = step & 1;
    const int kv0 = step * 32;

    wait_async0();    // buffer `cur` has landed (issued last iteration)
    __syncthreads();  // all waves past previous step (incl. P/V reads)

    if (step + 1 < SEQ / 32) issue(cur ^ 1, kv0 + 32);  // overlap next DMA

    // S = Q * K^T for a 16x32 block of scores
    v8f s0 = {}, s1 = {};
#pragma unroll
    for (int k8 = 0; k8 < 8; ++k8) {
      const int kB = k8 * 32;
      v16h b0 = load_b16(ldsK[cur], C_DIM, l15, kB, lane);
      v16h b1 = load_b16(ldsK[cur], C_DIM, 16 + l15, kB, lane);
      s0 = wmma16(qf[k8], b0, s0);
      s1 = wmma16(qf[k8], b1, s1);
    }
    __syncthreads();  // all waves done reading K[cur]; alias P onto it

    _Float16* Pw = ldsK[cur] + wIB * (16 * 32);  // wave-private 1 KB P tile

    // Online softmax: query row 8g+r lives in element r; key dim across the
    // 16 lanes of each half-wave, so reduce with xor masks 1,2,4,8.
#pragma unroll
    for (int r = 0; r < 8; ++r) {
      float loc = fmaxf(s0[r], s1[r]);
#pragma unroll
      for (int m = 1; m <= 8; m <<= 1) loc = fmaxf(loc, __shfl_xor(loc, m, 32));
      const float mn = fmaxf(mrow[r], loc);
      const float f  = __expf(mrow[r] - mn);
      const float p0 = __expf(s0[r] - mn);
      const float p1 = __expf(s1[r] - mn);
      float rs = p0 + p1;
#pragma unroll
      for (int m = 1; m <= 8; m <<= 1) rs += __shfl_xor(rs, m, 32);
      lrow[r] = lrow[r] * f + rs;
      mrow[r] = mn;
#pragma unroll
      for (int t = 0; t < 16; ++t) o[t][r] *= f;
      // stash probabilities (C-layout -> row-major 16x32 tile, wave-private)
      Pw[(8 * g + r) * 32 + l15]      = (_Float16)p0;
      Pw[(8 * g + r) * 32 + 16 + l15] = (_Float16)p1;
    }

    // O += P * V across all 16 channel tiles (V tile is channel-major in LDS)
    v16h pa = load_a16(Pw, 32, l15, 0, lane);
#pragma unroll
    for (int t = 0; t < 16; ++t) {
      v16h bv = load_b16(ldsV[cur], 32, 16 * t + l15, 0, lane);
      o[t] = wmma16(pa, bv, o[t]);
    }
  }

  // Normalize by row sums and write O in f16.
#pragma unroll
  for (int r = 0; r < 8; ++r) {
    const float inv = 1.0f / lrow[r];
    const size_t rowoff = (size_t)(qRow0 + 8 * g + r) * C_DIM;
#pragma unroll
    for (int t = 0; t < 16; ++t) {
      o16[rowoff + 16 * t + l15] = (_Float16)(o[t][r] * inv);
    }
  }
}

// ------------------------------------------- output projection + residual

__global__ void __launch_bounds__(256) proj_kernel(const _Float16* __restrict__ o16,
                                                   const _Float16* __restrict__ wTp,
                                                   const float* __restrict__ bp,
                                                   float* __restrict__ out) {
  const int lane = threadIdx.x & 31;
  const int l15 = lane & 15, g = lane >> 4;
  const int wid = (blockIdx.x * blockDim.x + threadIdx.x) >> 5;
  const int mTile = wid >> 2;
  const int nBase = (wid & 3) * 64;
  const int rowBase = mTile * 16;

  v8f acc[4] = {};
  for (int k8 = 0; k8 < 8; ++k8) {
    const int kB = k8 * 32;
    v16h a = load_a16(o16, C_DIM, rowBase + l15, kB, lane);
#pragma unroll
    for (int t = 0; t < 4; ++t) {
      v16h b = load_b16(wTp, C_DIM, nBase + 16 * t + l15, kB, lane);
      acc[t] = wmma16(a, b, acc[t]);
    }
  }
#pragma unroll
  for (int t = 0; t < 4; ++t) {
    const int col = nBase + 16 * t + l15;
    const float bcol = bp[col];
#pragma unroll
    for (int r = 0; r < 8; ++r) {
      const size_t idx = (size_t)(rowBase + 8 * g + r) * C_DIM + col;
      out[idx] += acc[t][r] + bcol;   // out already holds xn (residual)
    }
  }
}

// ---------------------------------------------------------------- launcher

extern "C" void kernel_launch(void* const* d_in, const int* in_sizes, int n_in,
                              void* d_out, int out_size, void* d_ws, size_t ws_size,
                              hipStream_t stream) {
  const float* x     = (const float*)d_in[0];
  const float* gamma = (const float*)d_in[1];
  const float* beta  = (const float*)d_in[2];
  const float* w_q   = (const float*)d_in[3];
  const float* b_q   = (const float*)d_in[4];
  const float* w_k   = (const float*)d_in[5];
  const float* b_k   = (const float*)d_in[6];
  const float* w_v   = (const float*)d_in[7];
  const float* b_v   = (const float*)d_in[8];
  const float* w_p   = (const float*)d_in[9];
  const float* b_p   = (const float*)d_in[10];
  float* out = (float*)d_out;

  const size_t TOKC = (size_t)NTOK * C_DIM;
  _Float16* x16  = (_Float16*)d_ws;
  _Float16* q16  = x16 + TOKC;
  _Float16* k16  = q16 + TOKC;
  _Float16* vT16 = k16 + TOKC;
  _Float16* o16  = vT16 + TOKC;
  _Float16* wTq  = o16 + TOKC;
  _Float16* wTk  = wTq + C_DIM * C_DIM;
  _Float16* wTv  = wTk + C_DIM * C_DIM;
  _Float16* wTp  = wTv + C_DIM * C_DIM;

  wt_kernel<<<256, 256, 0, stream>>>(w_q, wTq);
  wt_kernel<<<256, 256, 0, stream>>>(w_k, wTk);
  wt_kernel<<<256, 256, 0, stream>>>(w_v, wTv);
  wt_kernel<<<256, 256, 0, stream>>>(w_p, wTp);

  ln_kernel<<<NTOK / 8, 256, 0, stream>>>(x, gamma, beta, out, x16);

  qkv_kernel<<<(NTOK / 16) * 12 / 8, 256, 0, stream>>>(
      x16, wTq, wTk, wTv, b_q, b_k, b_v, q16, k16, vT16);

  attn_kernel<<<(NBATCH * SEQ / 16) / 4, 128, 0, stream>>>(q16, k16, vT16, o16);

  proj_kernel<<<(NTOK / 16) * 4 / 8, 256, 0, stream>>>(o16, wTp, b_p, out);
}
